// ScaledDotProductAttention_15229954031763
// MI455X (gfx1250) — matrix operand order repaired and checked by reading, not compiled, not run on hardware
//
#include <hip/hip_runtime.h>
#include <stdint.h>

#define S_LEN   2048
#define D_DIM   64
#define BM      64      // query rows per workgroup (16 per wave)
#define BN      64      // keys per LDS tile
#define KSTRIDE 72      // f16 row stride Kh  (144 B, 16B-aligned rows)
#define VSTRIDE 80      // f16 row stride VhT (160 B, 16B-aligned rows)

typedef __attribute__((ext_vector_type(16))) _Float16 v16h;
typedef __attribute__((ext_vector_type(4)))  _Float16 v4h;
typedef __attribute__((ext_vector_type(8)))  float    v8f;
typedef __attribute__((ext_vector_type(4)))  unsigned int v4u;
typedef __attribute__((ext_vector_type(8)))  int      v8i;
typedef __attribute__((ext_vector_type(4)))  int      v4i;

#if defined(__has_builtin)
#if __has_builtin(__builtin_amdgcn_tensor_load_to_lds) && __has_builtin(__builtin_amdgcn_s_wait_tensorcnt)
#define HAS_TDM 1
#endif
#if __has_builtin(__builtin_amdgcn_exp2f)
#define EXP2F(x) __builtin_amdgcn_exp2f(x)
#endif
#endif
#ifndef EXP2F
#define EXP2F(x) exp2f(x)
#endif

// ---------------------------------------------------------------------------
// TDM: load a 64x64 f16 tile into LDS with row padding (D# per ISA ch.8).
// t_d0/t_d1: tensor dims (data_size units), stride0: row stride,
// pad_amt_code: pad DWORDs-1 appended each 32 stored DWORDs (one 128B row).
// ---------------------------------------------------------------------------
#ifdef HAS_TDM
__device__ __forceinline__ void tdm_load_tile_f16(const _Float16* gtile,
                                                  unsigned lds_off, int t_d0,
                                                  int t_d1, int stride0,
                                                  int pad_amt_code) {
  const unsigned long long ga = (unsigned long long)(uintptr_t)gtile;
  v4u g0;
  g0[0] = 1u;                                        // count=1, user mode
  g0[1] = lds_off;                                   // lds_addr
  g0[2] = (unsigned)(ga & 0xFFFFFFFFull);            // global_addr[31:0]
  g0[3] = (unsigned)((ga >> 32) & 0x01FFFFFFull) | 0x80000000u; // +type=2
  v8i g1;
  g1[0] = (1 << 16) | (1 << 20) | (4 << 22) | (pad_amt_code << 25);
  //        data=2B    pad_en     interval=32DW      pad amount
  g1[1] = (t_d0 & 0xFFFF) << 16;                     // tensor_dim0 lo
  g1[2] = ((t_d0 >> 16) & 0xFFFF) | ((t_d1 & 0xFFFF) << 16);
  g1[3] = ((t_d1 >> 16) & 0xFFFF) | (64 << 16);      // tile_dim0 = 64
  g1[4] = 64;                                        // tile_dim1=64, dim2=0
  g1[5] = stride0;                                   // tensor_dim0_stride
  g1[6] = 0;
  g1[7] = 0;
  v4i gz4 = {0, 0, 0, 0};
#if __clang_major__ >= 23
  v8i gz8 = {0, 0, 0, 0, 0, 0, 0, 0};
  __builtin_amdgcn_tensor_load_to_lds(g0, g1, gz4, gz4, gz8, 0);
#else
  __builtin_amdgcn_tensor_load_to_lds(g0, g1, gz4, gz4, 0);
#endif
}
#endif

// ---------------------------------------------------------------------------
// One 32-key tile, transposed dataflow:
//   S^T(key,query) = K x Q^T ;  softmax per query = per lane ;  O^T = V^T x P
// ---------------------------------------------------------------------------
__device__ __forceinline__ void attn_tile_step(
    const _Float16 (*Kt)[KSTRIDE], const _Float16 (*Vt)[VSTRIDE],
    int n0rel, int n0, int q_abs, int lane16, int half, const v16h* qb,
    bool need_mask, float& m_run, float& l_run, v8f* acc) {
  // scores: two C-tiles (m = key-in-subtile, n = query)
  v8f st[2];
  st[0] = 0.0f; st[1] = 0.0f;
#pragma unroll
  for (int kc = 0; kc < 2; ++kc) {
#pragma unroll
    for (int t = 0; t < 2; ++t) {
      v16h ka;  // A = K rows (m=key, k=d chunk), contiguous A slot layout
      const _Float16* src = &Kt[n0rel + t * 16 + lane16][kc * 32 + 16 * half];
#pragma unroll
      for (int s = 0; s < 8; ++s) { ka[s] = src[s]; ka[s + 8] = src[8 + s]; }
      st[t] = __builtin_amdgcn_wmma_f32_16x16x32_f16(
          false, ka, false, qb[kc], (short)0, st[t], false, false);
    }
  }
  // causal mask (diagonal tiles only; scale already folded into Q)
  if (need_mask) {
#pragma unroll
    for (int t = 0; t < 2; ++t)
#pragma unroll
      for (int vi = 0; vi < 8; ++vi) {
        const int key = n0 + t * 16 + vi + 8 * half;
        if (key > q_abs) st[t][vi] = -1e30f;
      }
  }
  // per-query max: 15 in-lane maxes + one cross-half shuffle
  float x = fmaxf(st[0][0], st[1][0]);
#pragma unroll
  for (int vi = 1; vi < 8; ++vi) x = fmaxf(x, fmaxf(st[0][vi], st[1][vi]));
  x = fmaxf(x, __shfl_xor(x, 16));
  const float m_new = fmaxf(m_run, x);
  const float corr  = EXP2F(m_run - m_new);
  m_run = m_new;
  // p = 2^(s' - m') (log2e folded into Q scale); in-lane sum + cross-half
  float ssum = 0.0f;
#pragma unroll
  for (int t = 0; t < 2; ++t)
#pragma unroll
    for (int vi = 0; vi < 8; ++vi) {
      const float p = EXP2F(st[t][vi] - m_new);
      st[t][vi] = p;
      ssum += p;
    }
  ssum += __shfl_xor(ssum, 16);
  l_run = l_run * corr + ssum;
  // P B-fragment (k=key, n=query) straight from registers via half swap:
  // lane(n,h) needs st[h][0..7] from both halves (slots s<8 from half0 lane)
  v16h pfrag;
#pragma unroll
  for (int vi = 0; vi < 8; ++vi) {
    const float mine = half ? st[1][vi] : st[0][vi];
    const float give = half ? st[0][vi] : st[1][vi];
    const float oth  = __shfl_xor(give, 16);
    pfrag[vi]     = (_Float16)(half ? oth : mine);
    pfrag[vi + 8] = (_Float16)(half ? mine : oth);
  }
  // O^T accumulate: A = V^T (m=d, k=key), B = P
#pragma unroll
  for (int dt = 0; dt < 4; ++dt) {
#pragma unroll
    for (int vi = 0; vi < 8; ++vi) acc[dt][vi] *= corr;
    v16h va;
    const _Float16* src = &Vt[dt * 16 + lane16][n0rel + 16 * half];
#pragma unroll
    for (int s = 0; s < 8; ++s) { va[s] = src[s]; va[s + 8] = src[8 + s]; }
    acc[dt] = __builtin_amdgcn_wmma_f32_16x16x32_f16(
        false, va, false, pfrag, (short)0, acc[dt], false, false);
  }
}

// ---- shared prologue/epilogue helpers ------------------------------------
__device__ __forceinline__ void load_q_bfrag(const float* qrow, int half,
                                             v16h* qb) {
  const float QS = 0.125f * 1.44269504088896340736f;  // 1/sqrt(D) * log2(e)
#pragma unroll
  for (int kc = 0; kc < 2; ++kc)
#pragma unroll
    for (int s = 0; s < 16; ++s)
      qb[kc][s] = (_Float16)(qrow[kc * 32 + 16 * half + s] * QS);
}

__device__ __forceinline__ void store_out(float* op, int q_abs, int lane16,
                                          int half, float l_run, v8f* acc) {
  const float inv_l = 1.0f / l_run;
  float* orow = op + (size_t)q_abs * D_DIM;
#pragma unroll
  for (int dt = 0; dt < 4; ++dt) {
    float4 o0, o1;
    o0.x = acc[dt][0] * inv_l; o0.y = acc[dt][1] * inv_l;
    o0.z = acc[dt][2] * inv_l; o0.w = acc[dt][3] * inv_l;
    o1.x = acc[dt][4] * inv_l; o1.y = acc[dt][5] * inv_l;
    o1.z = acc[dt][6] * inv_l; o1.w = acc[dt][7] * inv_l;
    *(float4*)(orow + dt * 16 + 8 * half)     = o0;
    *(float4*)(orow + dt * 16 + 8 * half + 4) = o1;
  }
}

// ---------------------------------------------------------------------------
// Pre-pass: K -> f16 row-major [bh][S][64]; V -> f16 transposed [bh][64][S]
// ---------------------------------------------------------------------------
__global__ __launch_bounds__(256)
void cvt_kv_kernel(const float* __restrict__ k, const float* __restrict__ v,
                   _Float16* __restrict__ k16, _Float16* __restrict__ v16t) {
  __shared__ _Float16 T[64][72];
  const int tid = threadIdx.x;
  const int jb  = blockIdx.x * 64;
  const int bh  = blockIdx.y;
  const float*    kp = k   + (size_t)bh * S_LEN * D_DIM + (size_t)jb * D_DIM;
  const float*    vp = v   + (size_t)bh * S_LEN * D_DIM + (size_t)jb * D_DIM;
  _Float16*       ko = k16 + (size_t)bh * S_LEN * D_DIM + (size_t)jb * D_DIM;
  _Float16*       vo = v16t + (size_t)bh * D_DIM * S_LEN + jb;
#pragma unroll
  for (int i = 0; i < 4; ++i) {
    const int c = tid + i * 256;                      // 1024 float4 chunks
    const int row = c >> 4, col = (c & 15) << 2;
    const float4 k4 = *(const float4*)(kp + (row << 6) + col);
    v4h kh;
    kh[0] = (_Float16)k4.x; kh[1] = (_Float16)k4.y;
    kh[2] = (_Float16)k4.z; kh[3] = (_Float16)k4.w;
    *(v4h*)(ko + (row << 6) + col) = kh;
    const float4 v4 = *(const float4*)(vp + (row << 6) + col);
    T[row][col + 0] = (_Float16)v4.x; T[row][col + 1] = (_Float16)v4.y;
    T[row][col + 2] = (_Float16)v4.z; T[row][col + 3] = (_Float16)v4.w;
  }
  __syncthreads();
#pragma unroll
  for (int i = 0; i < 4; ++i) {
    const int c = tid + i * 256;
    const int d = c >> 4, kk = (c & 15) << 2;
    v4h o;
    o[0] = T[kk + 0][d]; o[1] = T[kk + 1][d];
    o[2] = T[kk + 2][d]; o[3] = T[kk + 3][d];
    *(v4h*)(vo + (size_t)d * S_LEN + kk) = o;
  }
}

// ---------------------------------------------------------------------------
// Main kernel, TDM-fed: tiles land in LDS already in final f16 layouts,
// double-buffered, next tile's DMA issued before compute (ASYNC overlap).
// ---------------------------------------------------------------------------
__global__ __launch_bounds__(128)
void fa_tdm_kernel(const float* __restrict__ q,
                   const _Float16* __restrict__ k16,
                   const _Float16* __restrict__ v16t,
                   float* __restrict__ out) {
  __shared__ _Float16 Kh[2][BN][KSTRIDE];
  __shared__ _Float16 VhT[2][D_DIM][VSTRIDE];

  const int tid = threadIdx.x, wave = tid >> 5, lane = tid & 31;
  const int lane16 = lane & 15, half = lane >> 4;
  const int bh = blockIdx.y, qbase = blockIdx.x * BM;
  const int q0 = qbase + wave * 16;
  const int q_abs = q0 + lane16;

  const float*    qp = q    + (size_t)bh * S_LEN * D_DIM;
  const _Float16* kb = k16  + (size_t)bh * S_LEN * D_DIM;
  const _Float16* vb = v16t + (size_t)bh * D_DIM * S_LEN;
  float*          op = out  + (size_t)bh * S_LEN * D_DIM;

  v16h qb2[2];
  load_q_bfrag(qp + (size_t)q_abs * D_DIM, half, qb2);

  float m_run = -1e30f, l_run = 0.0f;
  v8f acc[4];
#pragma unroll
  for (int dt = 0; dt < 4; ++dt) acc[dt] = 0.0f;

  const int kv_end = qbase + BM;                      // uniform causal limit

#ifdef HAS_TDM
  if (wave == 0) {                                    // prologue DMA for jb=0
    tdm_load_tile_f16(kb, (unsigned)(uintptr_t)&Kh[0][0][0],
                      D_DIM, S_LEN, D_DIM, 3);
    tdm_load_tile_f16(vb, (unsigned)(uintptr_t)&VhT[0][0][0],
                      S_LEN, D_DIM, S_LEN, 7);
  }
#endif
  int buf = 0;
  for (int jb = 0; jb < kv_end; jb += BN, buf ^= 1) {
#ifdef HAS_TDM
    if (wave == 0) __builtin_amdgcn_s_wait_tensorcnt(0);
    __syncthreads();                                  // tiles[buf] ready
    if (wave == 0 && jb + BN < kv_end) {              // overlap next DMA
      tdm_load_tile_f16(kb + (size_t)(jb + BN) * D_DIM,
                        (unsigned)(uintptr_t)&Kh[buf ^ 1][0][0],
                        D_DIM, S_LEN, D_DIM, 3);
      tdm_load_tile_f16(vb + (jb + BN),
                        (unsigned)(uintptr_t)&VhT[buf ^ 1][0][0],
                        S_LEN, D_DIM, S_LEN, 7);
    }
#else
    __syncthreads();
#pragma unroll
    for (int i = 0; i < 8; ++i) {                     // plain f16 tile copies
      const int c = tid + i * 128;
      const int r = c >> 4, cc = (c & 15) << 2;
      *(v4h*)&Kh[buf][r][cc]  = *(const v4h*)(kb + (size_t)(jb + r) * D_DIM + cc);
      *(v4h*)&VhT[buf][r][cc] = *(const v4h*)(vb + (size_t)r * S_LEN + jb + cc);
    }
    __syncthreads();
#endif
#pragma unroll
    for (int t = 0; t < 2; ++t) {
      const int n0 = jb + t * 32;
      if (n0 > q0 + 15) continue;                     // fully masked, uniform
      attn_tile_step(Kh[buf], VhT[buf], t * 32, n0, q_abs, lane16, half, qb2,
                     (n0 + 31 > q0), m_run, l_run, acc);
    }
  }
  store_out(op, q_abs, lane16, half, l_run, acc);
}

// ---------------------------------------------------------------------------
// Self-contained fallback (no workspace): f32 global -> cvt -> f16 LDS tiles
// ---------------------------------------------------------------------------
__global__ __launch_bounds__(128)
void fa_plain_kernel(const float* __restrict__ q, const float* __restrict__ k,
                     const float* __restrict__ v, float* __restrict__ out) {
  __shared__ _Float16 Kh[BN][KSTRIDE];
  __shared__ _Float16 VhT[D_DIM][VSTRIDE];

  const int tid = threadIdx.x, wave = tid >> 5, lane = tid & 31;
  const int lane16 = lane & 15, half = lane >> 4;
  const int bh = blockIdx.y, qbase = blockIdx.x * BM;
  const int q0 = qbase + wave * 16;
  const int q_abs = q0 + lane16;
  const size_t base = (size_t)bh * S_LEN * D_DIM;
  const float *qp = q + base, *kp = k + base, *vp = v + base;
  float* op = out + base;

  v16h qb2[2];
  load_q_bfrag(qp + (size_t)q_abs * D_DIM, half, qb2);

  float m_run = -1e30f, l_run = 0.0f;
  v8f acc[4];
#pragma unroll
  for (int dt = 0; dt < 4; ++dt) acc[dt] = 0.0f;

  const int kv_end = qbase + BM;
  for (int jb = 0; jb < kv_end; jb += BN) {
    __syncthreads();
#pragma unroll
    for (int i = 0; i < 8; ++i) {
      const int c = tid + i * 128;
      const int row = c >> 4, col = (c & 15) << 2;
      const float4 k4 = *(const float4*)(kp + (size_t)(jb + row) * D_DIM + col);
      v4h kh;
      kh[0] = (_Float16)k4.x; kh[1] = (_Float16)k4.y;
      kh[2] = (_Float16)k4.z; kh[3] = (_Float16)k4.w;
      *(v4h*)&Kh[row][col] = kh;
      const float4 v4 = *(const float4*)(vp + (size_t)(jb + row) * D_DIM + col);
      VhT[col + 0][row] = (_Float16)v4.x;
      VhT[col + 1][row] = (_Float16)v4.y;
      VhT[col + 2][row] = (_Float16)v4.z;
      VhT[col + 3][row] = (_Float16)v4.w;
    }
    if (jb + BN < kv_end) {
      __builtin_prefetch(kp + (size_t)(jb + BN) * D_DIM + tid * 32, 0, 1);
      __builtin_prefetch(vp + (size_t)(jb + BN) * D_DIM + tid * 32, 0, 1);
    }
    __syncthreads();
#pragma unroll
    for (int t = 0; t < 2; ++t) {
      const int n0 = jb + t * 32;
      if (n0 > q0 + 15) continue;
      attn_tile_step(Kh, VhT, t * 32, n0, q_abs, lane16, half, qb2,
                     (n0 + 31 > q0), m_run, l_run, acc);
    }
  }
  store_out(op, q_abs, lane16, half, l_run, acc);
}

extern "C" void kernel_launch(void* const* d_in, const int* in_sizes, int n_in,
                              void* d_out, int out_size, void* d_ws, size_t ws_size,
                              hipStream_t stream) {
  const float* q = (const float*)d_in[0];
  const float* k = (const float*)d_in[1];
  const float* v = (const float*)d_in[2];
  float* out = (float*)d_out;
  const int BH = in_sizes[0] / (S_LEN * D_DIM);       // B*H = 32

  const size_t tensor_elems = (size_t)BH * S_LEN * D_DIM;
  const size_t need = 2 * tensor_elems * sizeof(unsigned short);
  if (ws_size >= need) {
    _Float16* k16  = (_Float16*)d_ws;
    _Float16* v16t = k16 + tensor_elems;
    cvt_kv_kernel<<<dim3(S_LEN / 64, BH), 256, 0, stream>>>(k, v, k16, v16t);
    fa_tdm_kernel<<<dim3(S_LEN / BM, BH), 128, 0, stream>>>(q, k16, v16t, out);
  } else {
    fa_plain_kernel<<<dim3(S_LEN / BM, BH), 128, 0, stream>>>(q, k, v, out);
  }
}
